// GNN_38766374813707
// MI455X (gfx1250) — matrix operand order repaired and checked by reading, not compiled, not run on hardware
//
#include <hip/hip_runtime.h>
#include <hip/hip_bf16.h>

typedef __bf16 bf16_t;
typedef __attribute__((ext_vector_type(16))) __bf16 v16bf;
typedef __attribute__((ext_vector_type(8)))  __bf16 v8bf;
typedef __attribute__((ext_vector_type(8)))  float  v8f;

#define BB   32
#define NN   96
#define INF  64
#define HID  256
#define OUTF 128
#define KGC  3328          // (3*J-2)*HID = 13*256
#define ROWS (BB*NN)       // 3072

// ---- WMMA fragment loaders (layouts per CDNA5 ISA 7.12.2) ------------------
// A (16x32, bf16): lane m=L&15, hi=L>>4; elements 0..7 -> K = kbase+hi*8+e,
//                  elements 8..15 -> K = kbase+16+hi*8+(e-8).
__device__ __forceinline__ v16bf load_a_frag(const bf16_t* M, int ld, int row0,
                                             int kbase, int lane) {
  int m = lane & 15, hi = lane >> 4;
  const bf16_t* r = M + (size_t)(row0 + m) * ld + kbase + hi * 8;
  v8bf a0 = *(const v8bf*)(r);
  v8bf a1 = *(const v8bf*)(r + 16);
  v16bf out;
#pragma unroll
  for (int t = 0; t < 8; ++t) { out[t] = a0[t]; out[t + 8] = a1[t]; }
  return out;
}

// B (32x16, bf16): lane n=L&15, hi=L>>4; elements 0..7 -> K = kbase+hi*8+e,
// elements 8..15 -> K = kbase+16+hi*8+(e-8) of column col0+n.  Wt is the transposed weight, row-major [Nout, K] (K contiguous).
__device__ __forceinline__ v16bf load_b_frag(const bf16_t* Wt, int ld, int col0,
                                             int kbase, int lane) {
  int n = lane & 15, hi = lane >> 4;
  const bf16_t* r = Wt + (size_t)(col0 + n) * ld + kbase + hi * 8;
  v8bf b0 = *(const v8bf*)(r);
  v8bf b1 = *(const v8bf*)(r + 16);
  v16bf out;
#pragma unroll
  for (int t = 0; t < 8; ++t) { out[t] = b0[t]; out[t + 8] = b1[t]; }
  return out;
}

__device__ __forceinline__ v8f wmma_bf16f32(v16bf a, v16bf b, v8f c) {
  return __builtin_amdgcn_wmma_f32_16x16x32_bf16(false, a, false, b, (short)0, c,
                                                 false, false);
}

// ---- K0: transpose fp32 [K,G] -> bf16 [G,K] (K contiguous) -----------------
__global__ void __launch_bounds__(256)
transpose_bf16_kernel(const float* __restrict__ in, bf16_t* __restrict__ out,
                      int K, int G) {
  long long idx = (long long)blockIdx.x * 256 + threadIdx.x;
  long long total = (long long)K * G;
  if (idx >= total) return;
  int g = (int)(idx / K), k = (int)(idx % K);
  out[idx] = (bf16_t)in[(size_t)k * G + g];
}

// ---- K1: h = x @ W_emb ; also seed parts[:,0:256] with bf16(h) -------------
__global__ void __launch_bounds__(256)
emb_kernel(const float* __restrict__ x, const float* __restrict__ Wemb,
           float* __restrict__ h32, bf16_t* __restrict__ parts) {
  int r = blockIdx.x, g = threadIdx.x;
  const float* xr = x + (size_t)r * INF;
  float acc = 0.f;
#pragma unroll
  for (int k = 0; k < INF; ++k) acc += xr[k] * Wemb[(size_t)k * HID + g];
  h32[(size_t)r * HID + g] = acc;
  parts[(size_t)r * KGC + g] = (bf16_t)acc;
}

// ---- K2: hop adjacency masks ----------------------------------------------
__global__ void __launch_bounds__(256)
hop2_kernel(const float* __restrict__ Win, float* __restrict__ w2,
            unsigned char* __restrict__ m2) {
  int idx = blockIdx.x * 256 + threadIdx.x;      // 32*96*96 = 294912 exact
  int b = idx / (NN * NN), r = idx % (NN * NN);
  int i = r / NN, j = r % NN;
  const float* Wb = Win + (size_t)b * NN * NN;
  float acc = 0.f;
  for (int k = 0; k < NN; ++k) acc += Wb[i * NN + k] * Wb[k * NN + j];
  w2[idx] = acc;
  m2[idx] = (acc != 0.f);
}

__global__ void __launch_bounds__(256)
hop3_kernel(const float* __restrict__ w2, const float* __restrict__ Win,
            unsigned char* __restrict__ m3) {
  int idx = blockIdx.x * 256 + threadIdx.x;
  int b = idx / (NN * NN), r = idx % (NN * NN);
  int i = r / NN, j = r % NN;
  const float* W2b = w2 + (size_t)b * NN * NN;
  const float* Wb  = Win + (size_t)b * NN * NN;
  float acc = 0.f;
  for (int k = 0; k < NN; ++k) acc += W2b[i * NN + k] * Wb[k * NN + j];
  m3[idx] = (acc != 0.f);
}

// ---- K3: per (b,i,c): diff -> WMMA MLP (B panel held in VGPRs) -> agg ------
__global__ void __launch_bounds__(256)
edge_kernel(const float* __restrict__ h32, const float* __restrict__ Win,
            const unsigned char* __restrict__ m2,
            const unsigned char* __restrict__ m3,
            const bf16_t* __restrict__ W1t_all,
            const float* __restrict__ b1_0, const float* __restrict__ b1_1,
            const float* __restrict__ b1_2, const float* __restrict__ W2_0,
            const float* __restrict__ W2_1, const float* __restrict__ W2_2,
            const float* __restrict__ b2_0, const float* __restrict__ b2_1,
            const float* __restrict__ b2_2, bf16_t* __restrict__ parts) {
  __shared__ bf16_t s_diff[NN * HID];   // 48 KB
  __shared__ bf16_t s_e[NN * HID];      // 48 KB
  __shared__ float  s_E[NN * 4];        // 1.5 KB

  int bi = blockIdx.x;                  // 0..3071  (= b*96 + i)
  int c  = blockIdx.y;                  // operator 0..2
  int b = bi / NN, i = bi % NN;
  int tid = threadIdx.x;

  const float* b1 = (c == 0) ? b1_0 : (c == 1) ? b1_1 : b1_2;
  const float* W2 = (c == 0) ? W2_0 : (c == 1) ? W2_1 : W2_2;
  const float* b2 = (c == 0) ? b2_0 : (c == 1) ? b2_1 : b2_2;
  const bf16_t* W1t = W1t_all + (size_t)c * HID * HID;

  const float* hB   = h32 + (size_t)b * NN * HID;
  const float* hrow = hB + (size_t)i * HID;

  // diff[j,f] = |h[i,f] - h[j,f]|  (bf16 into LDS)
  for (int idx = tid; idx < NN * HID; idx += 256) {
    int j = idx >> 8, f = idx & 255;
    s_diff[idx] = (bf16_t)fabsf(hrow[f] - hB[(size_t)j * HID + f]);
  }
  __syncthreads();

  // e = relu(diff @ W1 + b1): M=96 (6 tiles), N=256 (16 tiles), K=256 (8 steps)
  // Each wave owns 2 N-tiles; the full B panel (8 k-step frags = 64 VGPRs) is
  // hoisted into registers and reused across all 6 M-tiles (6x less L2 traffic).
  int lane = tid & 31, wv = tid >> 5;
  {
    int n = lane & 15, hi = lane >> 4;
#pragma unroll 1
    for (int nt = wv * 2; nt < wv * 2 + 2; ++nt) {
      v16bf Bf[8];
#pragma unroll
      for (int ks = 0; ks < 8; ++ks)
        Bf[ks] = load_b_frag(W1t, HID, nt * 16, ks * 32, lane);
      int g = nt * 16 + n;
      float bias = b1[g];
#pragma unroll 1
      for (int mtile = 0; mtile < 6; ++mtile) {
        v8f acc = {};
#pragma unroll
        for (int ks = 0; ks < 8; ++ks) {
          v16bf a = load_a_frag(s_diff, HID, mtile * 16, ks * 32, lane);
          acc = wmma_bf16f32(a, Bf[ks], acc);
        }
        // C layout: lane col n, hi=L>>4; VGPR v -> row mtile*16 + hi*8 + v
#pragma unroll
        for (int v = 0; v < 8; ++v) {
          float e = acc[v] + bias;
          if (e < 0.f) e = 0.f;
          s_e[(mtile * 16 + hi * 8 + v) * HID + g] = (bf16_t)e;
        }
      }
    }
  }
  __syncthreads();

  // E[j,k] = mask(b,i,j) * (e[j,:] @ W2[:,k] + b2[k])
  size_t edgeidx = ((size_t)b * NN + i) * NN;
  for (int idx = tid; idx < NN * 4; idx += 256) {
    int j = idx >> 2, k = idx & 3;
    float acc = b2[k];
    const bf16_t* er = s_e + (size_t)j * HID;
    for (int f = 0; f < HID; ++f) acc += (float)er[f] * W2[f * 4 + k];
    bool mk;
    if (c == 0)      mk = (Win[edgeidx + j] != 0.f);
    else if (c == 1) mk = (m2[edgeidx + j] != 0);
    else             mk = (m3[edgeidx + j] != 0);
    s_E[idx] = mk ? acc : 0.f;
  }
  __syncthreads();

  // agg[k,f] = sum_j E[j,k] * h[b,j,f]  ->  parts[bi, 256 + c*1024 + k*256 + f]
  {
    int f = tid;
    float a0 = 0, a1 = 0, a2 = 0, a3 = 0;
    for (int j = 0; j < NN; ++j) {
      float hv = hB[(size_t)j * HID + f];
      a0 += s_E[j * 4 + 0] * hv;
      a1 += s_E[j * 4 + 1] * hv;
      a2 += s_E[j * 4 + 2] * hv;
      a3 += s_E[j * 4 + 3] * hv;
    }
    bf16_t* pr = parts + (size_t)bi * KGC + HID + c * 1024;
    pr[f]       = (bf16_t)a0;
    pr[256 + f] = (bf16_t)a1;
    pr[512 + f] = (bf16_t)a2;
    pr[768 + f] = (bf16_t)a3;
  }
}

// ---- K4: y = relu(parts @ W_gc + b_gc), M=3072, K=3328, N=256 --------------
// 2x2 register blocking: 4 accumulators, 2 A + 2 B frags per k-step, 4 WMMAs.
__global__ void __launch_bounds__(256)
gemm_gc_kernel(const bf16_t* __restrict__ A, const bf16_t* __restrict__ Bt,
               const float* __restrict__ bias, bf16_t* __restrict__ Y) {
  int lane = threadIdx.x & 31, wv = threadIdx.x >> 5;
  int T = blockIdx.x * 8 + wv;          // 0..767 wave-tasks (96 Mpair x 8 Npair)
  int mp = T >> 3, np = T & 7;
  int m0 = mp * 32, n0 = np * 32;
  v8f acc00 = {}, acc01 = {}, acc10 = {}, acc11 = {};
  for (int ks = 0; ks < KGC / 32; ++ks) {
    int kb = ks * 32;
    v16bf a0 = load_a_frag(A, KGC, m0, kb, lane);
    v16bf a1 = load_a_frag(A, KGC, m0 + 16, kb, lane);
    v16bf b0 = load_b_frag(Bt, KGC, n0, kb, lane);
    v16bf b1 = load_b_frag(Bt, KGC, n0 + 16, kb, lane);
    acc00 = wmma_bf16f32(a0, b0, acc00);
    acc01 = wmma_bf16f32(a0, b1, acc01);
    acc10 = wmma_bf16f32(a1, b0, acc10);
    acc11 = wmma_bf16f32(a1, b1, acc11);
  }
  int n = lane & 15, hi = lane >> 4;
  float bv0 = bias[n0 + n], bv1 = bias[n0 + 16 + n];
#pragma unroll
  for (int v = 0; v < 8; ++v) {
    int r0 = m0 + hi * 8 + v, r1 = m0 + 16 + hi * 8 + v;
    float e;
    e = acc00[v] + bv0; if (e < 0.f) e = 0.f;
    Y[(size_t)r0 * HID + n0 + n] = (bf16_t)e;
    e = acc01[v] + bv1; if (e < 0.f) e = 0.f;
    Y[(size_t)r0 * HID + n0 + 16 + n] = (bf16_t)e;
    e = acc10[v] + bv0; if (e < 0.f) e = 0.f;
    Y[(size_t)r1 * HID + n0 + n] = (bf16_t)e;
    e = acc11[v] + bv1; if (e < 0.f) e = 0.f;
    Y[(size_t)r1 * HID + n0 + 16 + n] = (bf16_t)e;
  }
}

// ---- K5: out = y @ W_last + b_last, M=3072, K=256, N=128 -------------------
__global__ void __launch_bounds__(256)
gemm_last_kernel(const bf16_t* __restrict__ A, const bf16_t* __restrict__ Bt,
                 const float* __restrict__ bias, float* __restrict__ Out) {
  int lane = threadIdx.x & 31, wv = threadIdx.x >> 5;
  int T = blockIdx.x * 8 + wv;          // 0..1535 tiles (192 M x 8 N)
  int mtile = T >> 3, ntile = T & 7;
  v8f acc = {};
#pragma unroll
  for (int ks = 0; ks < 8; ++ks) {
    v16bf a = load_a_frag(A, HID, mtile * 16, ks * 32, lane);
    v16bf b = load_b_frag(Bt, HID, ntile * 16, ks * 32, lane);
    acc = wmma_bf16f32(a, b, acc);
  }
  int n = lane & 15, hi = lane >> 4;
  int col = ntile * 16 + n;
  float bv = bias[col];
#pragma unroll
  for (int v = 0; v < 8; ++v) {
    int row = mtile * 16 + hi * 8 + v;
    Out[(size_t)row * OUTF + col] = acc[v] + bv;
  }
}

extern "C" void kernel_launch(void* const* d_in, const int* in_sizes, int n_in,
                              void* d_out, int out_size, void* d_ws,
                              size_t ws_size, hipStream_t stream) {
  (void)in_sizes; (void)n_in; (void)out_size; (void)ws_size;

  // setup_inputs() dict order:
  const float* x     = (const float*)d_in[0];
  const float* Win   = (const float*)d_in[1];
  /* d_in[2] g_size: pass-through, unused */
  const float* Wemb  = (const float*)d_in[3];
  const float* Wgc   = (const float*)d_in[4];
  const float* bgc   = (const float*)d_in[5];
  const float* Wlast = (const float*)d_in[6];
  const float* blast = (const float*)d_in[7];
  const float* w1p[3] = {(const float*)d_in[8],  (const float*)d_in[12], (const float*)d_in[16]};
  const float* b1p[3] = {(const float*)d_in[9],  (const float*)d_in[13], (const float*)d_in[17]};
  const float* w2p[3] = {(const float*)d_in[10], (const float*)d_in[14], (const float*)d_in[18]};
  const float* b2p[3] = {(const float*)d_in[11], (const float*)d_in[15], (const float*)d_in[19]};

  // workspace carve (all chunks 256B aligned)
  char* base = (char*)d_ws;
  size_t off = 0;
  auto carve = [&](size_t bytes) -> void* {
    void* p = base + off;
    off += (bytes + 255) & ~(size_t)255;
    return p;
  };
  bf16_t* W1t   = (bf16_t*)carve((size_t)3 * HID * HID * sizeof(bf16_t));
  bf16_t* Wgct  = (bf16_t*)carve((size_t)KGC * HID * sizeof(bf16_t));
  bf16_t* Wlt   = (bf16_t*)carve((size_t)OUTF * HID * sizeof(bf16_t));
  float*  h32   = (float*)carve((size_t)ROWS * HID * sizeof(float));
  bf16_t* parts = (bf16_t*)carve((size_t)ROWS * KGC * sizeof(bf16_t));
  float*  w2hop = (float*)carve((size_t)BB * NN * NN * sizeof(float));
  unsigned char* m2 = (unsigned char*)carve((size_t)BB * NN * NN);
  unsigned char* m3 = (unsigned char*)carve((size_t)BB * NN * NN);
  bf16_t* ybuf  = (bf16_t*)carve((size_t)ROWS * HID * sizeof(bf16_t));

  // K0: weight transposes to bf16 K-contiguous layouts
  for (int c = 0; c < 3; ++c)
    transpose_bf16_kernel<<<(HID * HID + 255) / 256, 256, 0, stream>>>(
        w1p[c], W1t + (size_t)c * HID * HID, HID, HID);
  transpose_bf16_kernel<<<((KGC * HID) + 255) / 256, 256, 0, stream>>>(
      Wgc, Wgct, KGC, HID);
  transpose_bf16_kernel<<<(HID * OUTF + 255) / 256, 256, 0, stream>>>(
      Wlast, Wlt, HID, OUTF);

  // K1: embedding
  emb_kernel<<<ROWS, 256, 0, stream>>>(x, Wemb, h32, parts);

  // K2: hop masks
  hop2_kernel<<<(BB * NN * NN) / 256, 256, 0, stream>>>(Win, w2hop, m2);
  hop3_kernel<<<(BB * NN * NN) / 256, 256, 0, stream>>>(w2hop, Win, m3);

  // K3: edge MLP + aggregation (dominant WMMA work)
  edge_kernel<<<dim3(ROWS, 3), 256, 0, stream>>>(
      h32, Win, m2, m3, W1t, b1p[0], b1p[1], b1p[2], w2p[0], w2p[1], w2p[2],
      b2p[0], b2p[1], b2p[2], parts);

  // K4: GConv GEMM (2x2 blocked): 768 wave-tasks / 8 waves = 96 blocks
  gemm_gc_kernel<<<96, 256, 0, stream>>>(parts, Wgct, bgc, ybuf);

  // K5: final linear: 1536 tiles / 8 waves = 192 blocks
  gemm_last_kernel<<<192, 256, 0, stream>>>(ybuf, Wlt, blast, (float*)d_out);
}
